// WindowAttention_33019708572345
// MI455X (gfx1250) — matrix operand order repaired and checked by reading, not compiled, not run on hardware
//
#include <hip/hip_runtime.h>

typedef _Float16 f16;
typedef __attribute__((ext_vector_type(16))) _Float16 v16h;
typedef __attribute__((ext_vector_type(8)))  float    v8f;
typedef __attribute__((ext_vector_type(4)))  _Float16 h4;

#define NWIN   2048      // 32 * 8 * 8 windows
#define TOK    49
#define TP     64        // padded tokens
#define CDIM   512
#define HD     64
#define NHEAD  8
#define MROWS  100352    // NWIN * TOK
#define KC     32        // K per WMMA step
#define NKC    (CDIM / KC)
#define SAP    56        // sA LDS row stride (halfs): 112B, 16B-aligned, bank step 28
#define SP     72        // attention LDS row stride (halfs): 144B, 16B-aligned, bank step 36

union FragU { v16h v; uint4 q[2]; };

// A fragment (16x32 f16). Lane L: m=L&15, half=L>>4.
// halfs [16h, 16h+16) -> VGPR0..7.
__device__ __forceinline__ v16h frag_a(const f16* base, int stride, int lane) {
    FragU f;
    const int m = lane & 15, half = lane >> 4;
    const f16* p = base + m * stride + 16 * half;
    f.q[0] = *(const uint4*)(p);
    f.q[1] = *(const uint4*)(p + 8);
    return f.v;
}

// B fragment (32x16 f16) from [n][k] layout (K contiguous per output column).
// Lane L: n=L&15, half=L>>4. halfs [16h, 16h+16) -> VGPR0..7.
__device__ __forceinline__ v16h frag_b(const f16* baseT, int stride, int lane) {
    FragU f;
    const int n = lane & 15, half = lane >> 4;
    const f16* p = baseT + n * stride + 16 * half;
    f.q[0] = *(const uint4*)(p);
    f.q[1] = *(const uint4*)(p + 8);
    return f.v;
}

#define WMMA_F16(a, b, c) \
    __builtin_amdgcn_wmma_f32_16x16x32_f16(false, (a), false, (b), (short)0, (c), false, false)

// ---------------------------------------------------------------------------
// Weight prep: f32 -> f16, K-contiguous transposed layouts (one-time, ~2 MB,
// stays resident in the 192 MB L2 for all GEMM blocks).
// ---------------------------------------------------------------------------
__global__ __launch_bounds__(256)
void prep_wqkv_kernel(const float* __restrict__ w_qkv, f16* __restrict__ wT) {
    const int idx = blockIdx.x * 256 + threadIdx.x;          // 512 * 1536
    const int col = idx % (3 * CDIM);
    const int k   = idx / (3 * CDIM);
    const int sec  = col >> 9;
    const int head = (col >> 6) & 7;
    const int d    = col & 63;
    wT[((size_t)(head * 192 + sec * 64 + d)) * CDIM + k] = (f16)w_qkv[idx];
}

__global__ __launch_bounds__(256)
void prep_wproj_kernel(const float* __restrict__ w_proj, f16* __restrict__ wpT) {
    const int idx = blockIdx.x * 256 + threadIdx.x;          // 512 * 512
    const int n = idx & 511;
    const int k = idx >> 9;
    wpT[(size_t)n * CDIM + k] = (f16)w_proj[idx];
}

// ---------------------------------------------------------------------------
// Kernel 1: fused QKV projection + window attention, one block per (win, head)
// 128 threads = 4 wave32. QKV phase: waves split N (3 tiles each, B frags
// direct from global wT, unique per wave). Attention phases: waves split M.
// Double-buffered A tile, one barrier per K step.
// ---------------------------------------------------------------------------
__global__ __launch_bounds__(128)
void win_attn_qkv_kernel(const float* __restrict__ x,
                         const f16* __restrict__ wT,
                         const float* __restrict__ b_qkv,
                         f16* __restrict__ attn_out) {
    __shared__ f16 sA[2][TP][SAP];      // x tile f16, double-buffered
    __shared__ f16 q_s[TP][SP];         // q * 1/8
    __shared__ f16 k_s[TP][SP];         // k  ([n][k] for QK^T B frags)
    __shared__ f16 vT_s[HD][SP];        // v^T ([n=hd][k=tok] for AV B frags)
    __shared__ f16 p_s[TP][SP];         // softmax probs
    __shared__ long long rowBase[TP];   // clamped: rows 49..63 -> row 48

    const int tid  = threadIdx.x;
    const int lane = tid & 31;
    const int wave = tid >> 5;
    const int laneN = lane & 15;
    const int half  = lane >> 4;

    const int win  = blockIdx.x >> 3;
    const int head = blockIdx.x & 7;
    const int b  = win >> 6;
    const int bl = win & 63;
    const int hb = bl >> 3;
    const int wb = bl & 7;

    if (tid < TP) {
        const int t = (tid < TOK) ? tid : (TOK - 1);   // clamp pad rows (finite data ok)
        const int it = t / 7, jt = t % 7;
        const int n_orig = (hb * 7 + it) * 56 + (wb * 7 + jt);
        rowBase[tid] = ((long long)b * 3136 + n_orig) * CDIM;
    }
    __syncthreads();

    // per-thread staging row bases in registers (slot s = tid + 128*it, row = s>>3)
    long long rbase[4];
    int  scol[4];
#pragma unroll
    for (int it = 0; it < 4; ++it) {
        const int s = tid + it * 128;
        rbase[it] = rowBase[s >> 3];
        scol[it]  = (s & 7) * 4;
    }
    f16* srow[4];
#pragma unroll
    for (int it = 0; it < 4; ++it) {
        const int s = tid + it * 128;
        srow[it] = &sA[0][s >> 3][(s & 7) * 4];        // buffer 0 target (buffer 1 = +TP*SAP)
    }

    const f16* wTh = wT + (size_t)head * 192 * CDIM;

    v8f acc[4][3];
    const v8f vzero = {0.f, 0.f, 0.f, 0.f, 0.f, 0.f, 0.f, 0.f};
#pragma unroll
    for (int mt = 0; mt < 4; ++mt)
#pragma unroll
        for (int j = 0; j < 3; ++j) acc[mt][j] = vzero;

    // prologue: stage chunk 0 into buffer 0
#pragma unroll
    for (int it = 0; it < 4; ++it) {
        const float4 xv = *(const float4*)(x + rbase[it] + scol[it]);
        *(h4*)(srow[it]) = (h4){(f16)xv.x, (f16)xv.y, (f16)xv.z, (f16)xv.w};
    }

    for (int kc = 0; kc < NKC; ++kc) {
        __syncthreads();                               // sA[kc&1] visible

        // prefetch next chunk into registers (overlaps WMMA block)
        float4 pf[4];
        if (kc < NKC - 1) {
#pragma unroll
            for (int it = 0; it < 4; ++it)
                pf[it] = *(const float4*)(x + rbase[it] + (kc + 1) * KC + scol[it]);
        }

        // B fragments for this wave's 3 N-tiles, direct from global wT
        v16h bf[3];
#pragma unroll
        for (int j = 0; j < 3; ++j) {
            const int row = (wave * 3 + j) * 16 + laneN;
            FragU f;
            const f16* p = wTh + (size_t)row * CDIM + kc * KC + 16 * half;
            f.q[0] = *(const uint4*)(p);
            f.q[1] = *(const uint4*)(p + 8);
            bf[j] = f.v;
        }

        const f16* cur = &sA[kc & 1][0][0];
#pragma unroll
        for (int mt = 0; mt < 4; ++mt) {
            const v16h af = frag_a(cur + mt * 16 * SAP, SAP, lane);
#pragma unroll
            for (int j = 0; j < 3; ++j)
                acc[mt][j] = WMMA_F16(af, bf[j], acc[mt][j]);
        }

        // stage next chunk into the other buffer
        if (kc < NKC - 1) {
            const int boff = ((kc + 1) & 1) * TP * SAP;
#pragma unroll
            for (int it = 0; it < 4; ++it)
                *(h4*)(srow[it] + boff) =
                    (h4){(f16)pf[it].x, (f16)pf[it].y, (f16)pf[it].z, (f16)pf[it].w};
        }
    }
    __syncthreads();

    // epilogue: bias, scale q, scatter to q/k/vT LDS (n = wave*48 + j*16 + laneN)
#pragma unroll
    for (int j = 0; j < 3; ++j) {
        const int n   = wave * 48 + j * 16 + laneN;
        const int sec = n >> 6;
        const int d   = n & 63;
        const float bias = b_qkv[sec * CDIM + head * HD + d];
#pragma unroll
        for (int mt = 0; mt < 4; ++mt) {
#pragma unroll
            for (int r = 0; r < 8; ++r) {
                const int m = mt * 16 + r + 8 * half;
                const float val = acc[mt][j][r] + bias;
                if (sec == 0)      q_s[m][d]  = (f16)(val * 0.125f);   // 1/sqrt(64)
                else if (sec == 1) k_s[m][d]  = (f16)val;
                else               vT_s[d][m] = (f16)val;
            }
        }
    }
    __syncthreads();

    // ---- scores = q @ k^T : waves split M (16 rows each) --------------------
    v8f sc[4];
#pragma unroll
    for (int i = 0; i < 4; ++i) sc[i] = vzero;
#pragma unroll
    for (int ks = 0; ks < HD; ks += KC) {
        const v16h af = frag_a(&q_s[wave * 16][ks], SP, lane);
#pragma unroll
        for (int nt = 0; nt < 4; ++nt) {
            const v16h bfr = frag_b(&k_s[nt * 16][ks], SP, lane);
            sc[nt] = WMMA_F16(af, bfr, sc[nt]);
        }
    }

    // ---- softmax over 49 valid columns --------------------------------------
#pragma unroll
    for (int r = 0; r < 8; ++r) {
        const int row = wave * 16 + r + 8 * half;
        float mx = -3.0e38f;
#pragma unroll
        for (int nt = 0; nt < 4; ++nt)
            if (nt * 16 + laneN < TOK) mx = fmaxf(mx, sc[nt][r]);
#pragma unroll
        for (int off = 1; off < 16; off <<= 1)
            mx = fmaxf(mx, __shfl_xor(mx, off, 16));
        float e[4]; float sum = 0.f;
#pragma unroll
        for (int nt = 0; nt < 4; ++nt) {
            e[nt] = (nt * 16 + laneN < TOK) ? __expf(sc[nt][r] - mx) : 0.f;
            sum += e[nt];
        }
#pragma unroll
        for (int off = 1; off < 16; off <<= 1)
            sum += __shfl_xor(sum, off, 16);
        const float inv = __frcp_rn(sum);
#pragma unroll
        for (int nt = 0; nt < 4; ++nt)
            p_s[row][nt * 16 + laneN] = (f16)(e[nt] * inv);
    }
    __syncthreads();

    // ---- out = probs @ v ----------------------------------------------------
    v8f ov[4];
#pragma unroll
    for (int i = 0; i < 4; ++i) ov[i] = vzero;
#pragma unroll
    for (int ks = 0; ks < TP; ks += KC) {
        const v16h af = frag_a(&p_s[wave * 16][ks], SP, lane);
#pragma unroll
        for (int nt = 0; nt < 4; ++nt) {
            const v16h bfr = frag_b(&vT_s[nt * 16][ks], SP, lane);
            ov[nt] = WMMA_F16(af, bfr, ov[nt]);
        }
    }

#pragma unroll
    for (int nt = 0; nt < 4; ++nt) {
#pragma unroll
        for (int r = 0; r < 8; ++r) {
            const int m = wave * 16 + r + 8 * half;
            if (m < TOK) {
                const size_t g = (size_t)win * TOK + m;
                attn_out[g * CDIM + head * HD + nt * 16 + laneN] = (f16)ov[nt][r];
            }
        }
    }
}

// ---------------------------------------------------------------------------
// Kernel 2: projection [100352 x 512] @ [512 x 512] + bias + inverse window
// permutation. Block tile 64x64; waves split N (unique global B frags).
// Double-buffered A tile, one barrier per K step.
// ---------------------------------------------------------------------------
__global__ __launch_bounds__(128)
void win_attn_proj_kernel(const f16* __restrict__ attn_out,
                          const f16* __restrict__ wpT,
                          const float* __restrict__ b_proj,
                          float* __restrict__ out) {
    __shared__ f16 sA[2][64][SAP];

    const int tid  = threadIdx.x;
    const int lane = tid & 31;
    const int wave = tid >> 5;
    const int laneN = lane & 15;
    const int half  = lane >> 4;

    const int row0 = blockIdx.x * 64;
    const int n0   = blockIdx.y * 64;

    v8f acc[4];
    const v8f vzero = {0.f, 0.f, 0.f, 0.f, 0.f, 0.f, 0.f, 0.f};
#pragma unroll
    for (int i = 0; i < 4; ++i) acc[i] = vzero;

    const int colb = n0 + wave * 16;

    // per-thread copy slots: s = tid + 128*it, row = s>>2, chunk = (s&3)*8 halfs
    const f16* gsrc[2];
    f16* ldst[2];
#pragma unroll
    for (int it = 0; it < 2; ++it) {
        const int s = tid + it * 128;
        gsrc[it] = attn_out + (size_t)(row0 + (s >> 2)) * CDIM + (s & 3) * 8;
        ldst[it] = &sA[0][s >> 2][(s & 3) * 8];
    }

    // prologue: stage chunk 0
#pragma unroll
    for (int it = 0; it < 2; ++it)
        *(uint4*)(ldst[it]) = *(const uint4*)(gsrc[it]);

    for (int kc = 0; kc < NKC; ++kc) {
        __syncthreads();

        uint4 pf[2];
        if (kc < NKC - 1) {
#pragma unroll
            for (int it = 0; it < 2; ++it)
                pf[it] = *(const uint4*)(gsrc[it] + (kc + 1) * KC);
        }

        FragU f;
        const f16* p = wpT + (size_t)(colb + laneN) * CDIM + kc * KC + 16 * half;
        f.q[0] = *(const uint4*)(p);
        f.q[1] = *(const uint4*)(p + 8);
        const v16h bf = f.v;

        const f16* cur = &sA[kc & 1][0][0];
#pragma unroll
        for (int mt = 0; mt < 4; ++mt) {
            const v16h af = frag_a(cur + mt * 16 * SAP, SAP, lane);
            acc[mt] = WMMA_F16(af, bf, acc[mt]);
        }

        if (kc < NKC - 1) {
            const int boff = ((kc + 1) & 1) * 64 * SAP;
#pragma unroll
            for (int it = 0; it < 2; ++it)
                *(uint4*)(ldst[it] + boff) = pf[it];
        }
    }

    const int col = colb + laneN;
    const float bias = b_proj[col];
#pragma unroll
    for (int mt = 0; mt < 4; ++mt) {
#pragma unroll
        for (int r = 0; r < 8; ++r) {
            const unsigned g = row0 + mt * 16 + r + 8 * half;
            const unsigned win = g / TOK;
            const unsigned t   = g - win * TOK;
            const unsigned b   = win >> 6;
            const unsigned bl  = win & 63;
            const unsigned hb = bl >> 3, wb = bl & 7;
            const unsigned it = t / 7, jt = t % 7;
            const unsigned n_orig = (hb * 7 + it) * 56 + (wb * 7 + jt);
            out[((size_t)b * 3136 + n_orig) * CDIM + col] = acc[mt][r] + bias;
        }
    }
}

extern "C" void kernel_launch(void* const* d_in, const int* in_sizes, int n_in,
                              void* d_out, int out_size, void* d_ws, size_t ws_size,
                              hipStream_t stream) {
    const float* x      = (const float*)d_in[0];
    const float* w_qkv  = (const float*)d_in[1];
    const float* b_qkv  = (const float*)d_in[2];
    const float* w_proj = (const float*)d_in[3];
    const float* b_proj = (const float*)d_in[4];
    float* out = (float*)d_out;

    f16* wT   = (f16*)d_ws;                         // 8*192*512 halfs = 1.5 MB
    f16* wpT  = wT + (size_t)NHEAD * 192 * CDIM;    // 512*512 halfs   = 0.5 MB
    f16* attn = wpT + (size_t)CDIM * CDIM;          // 100352*512 halfs ~103 MB

    prep_wqkv_kernel<<<dim3((CDIM * 3 * CDIM) / 256), dim3(256), 0, stream>>>(w_qkv, wT);
    prep_wproj_kernel<<<dim3((CDIM * CDIM) / 256), dim3(256), 0, stream>>>(w_proj, wpT);

    win_attn_qkv_kernel<<<dim3(NWIN * NHEAD), dim3(128), 0, stream>>>(
        x, wT, b_qkv, attn);

    win_attn_proj_kernel<<<dim3(MROWS / 64, CDIM / 64), dim3(128), 0, stream>>>(
        attn, wpT, b_proj, out);
}
